// SelfAttention_73890617360613
// MI455X (gfx1250) — matrix operand order repaired and checked by reading, not compiled, not run on hardware
//
#include <hip/hip_runtime.h>
#include <hip/hip_bf16.h>

// Sizes fixed by the reference problem.
#define B_    2
#define T_    2048
#define D_    1024
#define H_    16
#define DH_   64
#define TOPK_ 64

typedef __bf16 bf16;
typedef __attribute__((ext_vector_type(16))) __bf16        v16bf;
typedef __attribute__((ext_vector_type(8)))  float         v8f;
typedef __attribute__((ext_vector_type(4)))  unsigned int  u32x4;
typedef __attribute__((ext_vector_type(8)))  unsigned int  u32x8;

#define NEGF  (-3.0e38f)
#define NEGH  (-1.0e38f)

union FragAB {            // 16 bf16 per lane = one WMMA A or B operand
  v16bf v;
  u32x4 u[2];
};

__device__ __forceinline__ v8f wmma_bf16(v16bf a, v16bf b, v8f c) {
  // D = A(16x32) * B(32x16) + C, f32 accum
  return __builtin_amdgcn_wmma_f32_16x16x32_bf16(
      false, a, false, b, (short)0, c, false, false);
}

__device__ __forceinline__ int wredsum_i(int v) {
  #pragma unroll
  for (int o = 16; o; o >>= 1) v += __shfl_xor(v, o, 32);
  return v;
}
__device__ __forceinline__ float wredsum_f(float v) {
  #pragma unroll
  for (int o = 16; o; o >>= 1) v += __shfl_xor(v, o, 32);
  return v;
}
__device__ __forceinline__ float wredmax_f(float v) {
  #pragma unroll
  for (int o = 16; o; o >>= 1) v = fmaxf(v, __shfl_xor(v, o, 32));
  return v;
}
// order-preserving float -> uint map (for exact top-k threshold search)
__device__ __forceinline__ unsigned int fmono(float f) {
  unsigned int u = __float_as_uint(f);
  return u ^ ((u & 0x80000000u) ? 0xFFFFFFFFu : 0x80000000u);
}

// ---------------------------------------------------------------------------
// Kernel 1: QKV projection.  out = x @ W^T + b, bf16.
//   z=0 -> Q in (B,H,T,DH);  z=1 -> K^T in (B,H,DH,T);  z=2 -> V in (B,H,T,DH)
// WG = 256 thr (8 waves), tile 128x128, wave tile 64x32 (4x2 WMMA tiles).
// ---------------------------------------------------------------------------
__global__ __launch_bounds__(256) void qkv_gemm(
    const float* __restrict__ x,
    const float* __restrict__ W0, const float* __restrict__ bias0,
    const float* __restrict__ W1, const float* __restrict__ bias1,
    const float* __restrict__ W2, const float* __restrict__ bias2,
    bf16* __restrict__ Qo, bf16* __restrict__ KTo, bf16* __restrict__ Vo) {
  __shared__ __align__(16) bf16 At[128][40];   // A tile, bf16, padded stride
  __shared__ __align__(16) bf16 Bt[32][136];   // B tile, K-major (transposed)

  const int tid  = threadIdx.x;
  const int lane = tid & 31;
  const int wid  = tid >> 5;
  const int rowBase = blockIdx.x * 128;        // over B*T = 4096 rows
  const int colBase = blockIdx.y * 128;        // over D = 1024 cols
  const int z = blockIdx.z;
  const float* W    = (z == 0) ? W0 : ((z == 1) ? W1 : W2);
  const float* bias = (z == 0) ? bias0 : ((z == 1) ? bias1 : bias2);
  bf16* out         = (z == 0) ? Qo : ((z == 1) ? KTo : Vo);

  const int mrow = (wid >> 2) * 64;            // 0 / 64
  const int ncol = (wid & 3) * 32;             // 0..96

  v8f acc[4][2];
  #pragma unroll
  for (int i = 0; i < 4; ++i)
    #pragma unroll
    for (int j = 0; j < 2; ++j) acc[i][j] = v8f{};

  for (int ks = 0; ks < 32; ++ks) {
    const int k0 = ks * 32;
    // A: 128x32 f32 -> bf16 LDS (row-major)
    #pragma unroll
    for (int rep = 0; rep < 4; ++rep) {
      int idx = tid + rep * 256;               // 0..1023 chunks of 4 floats
      int r = idx >> 3, c4 = (idx & 7) * 4;
      const float* src = &x[(size_t)(rowBase + r) * D_ + k0 + c4];
      const float4 f = *(const float4*)src;
      if (ks < 31) __builtin_prefetch(src + 32, 0, 0);   // next k-step tile
      bf16* d = &At[r][c4];
      d[0] = (bf16)f.x; d[1] = (bf16)f.y; d[2] = (bf16)f.z; d[3] = (bf16)f.w;
    }
    // B: 128 rows of W (contiguous in k) -> transposed K-major LDS
    #pragma unroll
    for (int rep = 0; rep < 4; ++rep) {
      int idx = tid + rep * 256;
      int r = idx >> 3, c4 = (idx & 7) * 4;
      const float* src = &W[(size_t)(colBase + r) * D_ + k0 + c4];
      const float4 f = *(const float4*)src;
      if (ks < 31) __builtin_prefetch(src + 32, 0, 0);
      Bt[c4 + 0][r] = (bf16)f.x; Bt[c4 + 1][r] = (bf16)f.y;
      Bt[c4 + 2][r] = (bf16)f.z; Bt[c4 + 3][r] = (bf16)f.w;
    }
    __syncthreads();

    const int m16 = lane & 15, hl = lane >> 4;
    FragAB a[4], bfr[2];
    #pragma unroll
    for (int i = 0; i < 4; ++i) {
      const bf16* ap = &At[mrow + i * 16 + m16][0];
      a[i].u[0] = *(const u32x4*)(ap + hl * 8);        // K = hl*8 .. +7
      a[i].u[1] = *(const u32x4*)(ap + 16 + hl * 8);   // K = 16+hl*8 .. +7
    }
    #pragma unroll
    for (int j = 0; j < 2; ++j) {
      const bf16* bp = &Bt[hl * 8][ncol + j * 16 + m16];  // N = m16, K = table order
      #pragma unroll
      for (int e = 0; e < 8; ++e) {
        bfr[j].v[e] = bp[e * 136];
        bfr[j].v[8 + e] = bp[(16 + e) * 136];
      }
    }
    #pragma unroll
    for (int i = 0; i < 4; ++i)
      #pragma unroll
      for (int j = 0; j < 2; ++j)
        acc[i][j] = wmma_bf16(a[i].v, bfr[j].v, acc[i][j]);
    __syncthreads();
  }

  // store (K goes out transposed)
  const int m16 = lane & 15, hl = lane >> 4;
  #pragma unroll
  for (int j = 0; j < 2; ++j) {
    const int c = colBase + ncol + j * 16 + m16;
    const float bv = bias[c];
    const int h = c >> 6, dh = c & 63;
    #pragma unroll
    for (int i = 0; i < 4; ++i) {
      #pragma unroll
      for (int r = 0; r < 8; ++r) {
        const int rowG = rowBase + mrow + i * 16 + r + hl * 8;
        const int bb = rowG >> 11, t = rowG & (T_ - 1);
        const bf16 val = (bf16)(acc[i][j][r] + bv);
        if (z == 1)
          out[(((size_t)(bb * H_ + h)) * DH_ + dh) * T_ + t] = val;  // K^T
        else
          out[(((size_t)(bb * H_ + h)) * T_ + t) * DH_ + dh] = val;  // Q / V
      }
    }
  }
}

// ---------------------------------------------------------------------------
// Kernel 2: attention with exact top-64 sparsification.
// One WG = (b, h, 16-query tile). 8 waves. Dynamic LDS:
//   Slds[16][2048] f32 logits->probs (128KB) | Opart 4KB
// K^T / V B-fragments come straight from global (ISA lane=K layout).
// ---------------------------------------------------------------------------
__global__ __launch_bounds__(256) void attn_topk(
    const bf16* __restrict__ Qg, const bf16* __restrict__ KT,
    const bf16* __restrict__ Vg, const unsigned char* __restrict__ pad,
    const int* __restrict__ causal_p, bf16* __restrict__ CTX) {
  extern __shared__ char smem[];
  float* Slds  = (float*)smem;                           // [16][2048]
  float* Opart = (float*)(smem + 16 * 2048 * 4);         // [4][8][32]

  const int tid = threadIdx.x, lane = tid & 31, wid = tid >> 5;
  const int q0 = blockIdx.x * 16;
  const int h = blockIdx.y, bb = blockIdx.z;
  const int causal = causal_p[0];
  const size_t headBase = ((size_t)(bb * H_ + h)) * T_ * DH_;  // Q/V base
  const int m16 = lane & 15, hl = lane >> 4;

  // init logits to -inf (covers causally skipped key tiles)
  for (int i = tid; i < 16 * 2048; i += 256) Slds[i] = NEGF;

  // Q fragments (A operand), straight from global bf16
  const bf16* qrow = Qg + headBase + (size_t)(q0 + m16) * DH_;
  FragAB aq[2];
  aq[0].u[0] = *(const u32x4*)(qrow + hl * 8);
  aq[0].u[1] = *(const u32x4*)(qrow + 16 + hl * 8);
  aq[1].u[0] = *(const u32x4*)(qrow + 32 + hl * 8);
  aq[1].u[1] = *(const u32x4*)(qrow + 48 + hl * 8);

  __syncthreads();  // init complete before S writes

  // S = Q K^T: each wave strides over its 16-key column tiles
  const float scale = 0.125f;                            // DH^-0.5
  const int jlimit = causal ? (q0 + 16) : T_;
  for (int j = wid * 16; j < jlimit; j += 128) {
    FragAB b0, b1;
    const bf16* kp0 = KT + headBase + (size_t)(hl * 8) * T_ + j + m16;  // key = j+m16, d table order
    const bf16* kp1 = kp0 + (size_t)32 * T_;                           // d += 32
    #pragma unroll
    for (int e = 0; e < 8; ++e) {
      b0.v[e] = kp0[(size_t)e * T_];
      b0.v[8 + e] = kp0[(size_t)(16 + e) * T_];
      b1.v[e] = kp1[(size_t)e * T_];
      b1.v[8 + e] = kp1[(size_t)(16 + e) * T_];
    }
    __builtin_prefetch(kp0 + 128, 0, 0);                 // next column tile
    __builtin_prefetch(kp1 + 128, 0, 0);
    v8f acc = v8f{};
    acc = wmma_bf16(aq[0].v, b0.v, acc);
    acc = wmma_bf16(aq[1].v, b1.v, acc);
    #pragma unroll
    for (int r = 0; r < 8; ++r) {
      const int m = r + hl * 8;
      const int kg = j + m16, qg = q0 + m;
      float val = acc[r] * scale;
      if ((causal && kg > qg) || pad[bb * T_ + kg]) val = NEGF;
      Slds[m * 2048 + kg] = val;
    }
  }
  __syncthreads();

  // exact top-64 per row + softmax (wave handles 2 rows)
  #pragma unroll
  for (int rr = 0; rr < 2; ++rr) {
    const int row = wid * 2 + rr;
    float* Lr = &Slds[row * 2048];
    const int limc = q0 + row + 1;
    const int lim = causal ? (limc < 2048 ? limc : 2048) : 2048;

    int cnt = 0;
    for (int i = lane; i < lim; i += 32) cnt += (Lr[i] > NEGH) ? 1 : 0;
    cnt = wredsum_i(cnt);

    unsigned int thr = 0u;
    if (cnt > TOPK_) {  // binary search on monotone uint for 64th largest
      unsigned int lo = 0u, hi = 0xFFFFFFFFu;
      for (int it = 0; it < 32; ++it) {
        const unsigned int d = hi - lo;
        const unsigned int mid = lo + (d >> 1) + (d & 1u);
        int c = 0;
        for (int i = lane; i < lim; i += 32) c += (fmono(Lr[i]) >= mid) ? 1 : 0;
        c = wredsum_i(c);
        if (c >= TOPK_) lo = mid; else hi = mid - 1u;
      }
      thr = lo;
    }
    float mx = NEGF;
    for (int i = lane; i < lim; i += 32) mx = fmaxf(mx, Lr[i]);
    mx = wredmax_f(mx);
    float sum = 0.f;
    for (int i = lane; i < lim; i += 32) {
      const float f = Lr[i];
      if (f > NEGH && fmono(f) >= thr) sum += __expf(f - mx);
    }
    sum = wredsum_f(sum);
    const float inv = (sum > 0.f) ? 1.f / sum : 0.f;
    for (int i = lane; i < 2048; i += 32) {
      const float f = Lr[i];
      float p = 0.f;
      if (i < lim && f > NEGH && fmono(f) >= thr) p = __expf(f - mx) * inv;
      Lr[i] = p;
    }
  }
  __syncthreads();

  // O = P x V : wave -> (col tile ct, key half); V B-frags straight from global
  {
    int limStep = 64;
    if (causal) {
      const int ls = (q0 + 16 + 31) >> 5;
      limStep = ls < 64 ? ls : 64;
    }
    const int ct = wid & 3, half = wid >> 2;
    v8f acc = v8f{};
    for (int s = half; s < limStep; s += 2) {
      const int kbase = s * 32;
      FragAB pa;
      const float* p0 = &Slds[m16 * 2048 + kbase + hl * 8];
      const float* p1 = &Slds[m16 * 2048 + kbase + 16 + hl * 8];
      #pragma unroll
      for (int e = 0; e < 8; ++e) pa.v[e] = (bf16)p0[e];
      #pragma unroll
      for (int e = 0; e < 8; ++e) pa.v[8 + e] = (bf16)p1[e];
      FragAB vb;
      const bf16* vp = Vg + headBase + (size_t)(kbase + hl * 8) * DH_ + ct * 16 + m16;
      #pragma unroll
      for (int e = 0; e < 8; ++e) {
        vb.v[e] = vp[e * DH_];
        vb.v[8 + e] = vp[(16 + e) * DH_];
      }
      acc = wmma_bf16(pa.v, vb.v, acc);
    }
    if (half == 1) {
      #pragma unroll
      for (int r = 0; r < 8; ++r) Opart[(ct * 8 + r) * 32 + lane] = acc[r];
    }
    __syncthreads();
    if (half == 0) {
      #pragma unroll
      for (int r = 0; r < 8; ++r) {
        const float v = acc[r] + Opart[(ct * 8 + r) * 32 + lane];
        const int m = r + hl * 8;
        const int t = q0 + m;
        const int c = h * DH_ + ct * 16 + m16;
        CTX[((size_t)(bb * T_ + t)) * D_ + c] = (bf16)v;
      }
    }
  }
}

// ---------------------------------------------------------------------------
// Kernel 3: output projection.  out = ctx @ Wo^T + bo  (f32 output)
// A tile (pure bf16 copy) is moved global->LDS by the Tensor Data Mover:
// D# describes a 32x128 tile of the (4096,1024) bf16 tensor with LDS padding
// (pad 4 DWORDs every 16 DWORDs -> padded stride 40 elems = At row stride).
// ---------------------------------------------------------------------------
__global__ __launch_bounds__(256) void out_gemm(
    const bf16* __restrict__ ctx, const float* __restrict__ Wo,
    const float* __restrict__ bo, float* __restrict__ out) {
  __shared__ __align__(16) bf16 At[128][40];
  __shared__ __align__(16) bf16 Bt[32][136];

  const int tid = threadIdx.x, lane = tid & 31, wid = tid >> 5;
  const int rowBase = blockIdx.x * 128;
  const int colBase = blockIdx.y * 128;
  const int mrow = (wid >> 2) * 64;
  const int ncol = (wid & 3) * 32;

  v8f acc[4][2];
  #pragma unroll
  for (int i = 0; i < 4; ++i)
    #pragma unroll
    for (int j = 0; j < 2; ++j) acc[i][j] = v8f{};

  for (int ks = 0; ks < 32; ++ks) {
    const int k0 = ks * 32;

    // A tile via TDM (wave 0 only issues; TDM ignores EXEC, one DMA per WG)
    if (wid == 0) {
      const unsigned long long ga = (unsigned long long)(const void*)(
          ctx + (size_t)rowBase * D_ + k0);
      const unsigned ldsa = (unsigned)(unsigned long long)(const void*)&At[0][0];
      u32x4 g0;
      g0[0] = 1u;                                  // count=1, user mode
      g0[1] = ldsa;                                // lds_addr
      g0[2] = (unsigned)ga;                        // global_addr[31:0]
      g0[3] = ((unsigned)(ga >> 32) & 0x01FFFFFFu) | (2u << 30);  // type=2
      u32x8 g1;
      g1[0] = (1u << 16)        // data_size = 2 bytes
            | (1u << 20)        // pad_enable
            | (3u << 22)        // pad_interval: every 16 DWORDs (64B)
            | (3u << 25);       // pad_amount: 4 DWORDs (16B) -> stride 40 elem
      g1[1] = (unsigned)D_ << 16;                  // tensor_dim0 = 1024 (lo16)
      g1[2] = ((unsigned)(B_ * T_) & 0xFFFFu) << 16;  // tensor_dim1 lo16
      g1[3] = 32u << 16;                           // tile_dim0 = 32
      g1[4] = 128u;                                // tile_dim1 = 128
      g1[5] = (unsigned)D_;                        // tensor_dim0_stride lo32
      g1[6] = 0u;
      g1[7] = 0u;
      asm volatile("tensor_load_to_lds %0, %1" :: "s"(g0), "s"(g1) : "memory");
    }

    // B tile: 128 rows of Wo f32 -> bf16, transposed K-major (all waves)
    #pragma unroll
    for (int rep = 0; rep < 4; ++rep) {
      int idx = tid + rep * 256;
      int r = idx >> 3, c4 = (idx & 7) * 4;
      const float* src = &Wo[(size_t)(colBase + r) * D_ + k0 + c4];
      const float4 f = *(const float4*)src;
      if (ks < 31) __builtin_prefetch(src + 32, 0, 0);
      Bt[c4 + 0][r] = (bf16)f.x; Bt[c4 + 1][r] = (bf16)f.y;
      Bt[c4 + 2][r] = (bf16)f.z; Bt[c4 + 3][r] = (bf16)f.w;
    }

    if (wid == 0) __builtin_amdgcn_s_wait_tensorcnt(0);  // A tile landed
    __syncthreads();

    const int m16 = lane & 15, hl = lane >> 4;
    FragAB a[4], bfr[2];
    #pragma unroll
    for (int i = 0; i < 4; ++i) {
      const bf16* ap = &At[mrow + i * 16 + m16][0];
      a[i].u[0] = *(const u32x4*)(ap + hl * 8);
      a[i].u[1] = *(const u32x4*)(ap + 16 + hl * 8);
    }
    #pragma unroll
    for (int j = 0; j < 2; ++j) {
      const bf16* bp = &Bt[hl * 8][ncol + j * 16 + m16];
      #pragma unroll
      for (int e = 0; e < 8; ++e) {
        bfr[j].v[e] = bp[e * 136];
        bfr[j].v[8 + e] = bp[(16 + e) * 136];
      }
    }
    #pragma unroll
    for (int i = 0; i < 4; ++i)
      #pragma unroll
      for (int j = 0; j < 2; ++j)
        acc[i][j] = wmma_bf16(a[i].v, bfr[j].v, acc[i][j]);
    __syncthreads();
  }

  const int m16 = lane & 15, hl = lane >> 4;
  #pragma unroll
  for (int j = 0; j < 2; ++j) {
    const int c = colBase + ncol + j * 16 + m16;
    const float bv = bo[c];
    #pragma unroll
    for (int i = 0; i < 4; ++i) {
      #pragma unroll
      for (int r = 0; r < 8; ++r) {
        const int rowG = rowBase + mrow + i * 16 + r + hl * 8;
        out[(size_t)rowG * D_ + c] = acc[i][j][r] + bv;
      }
    }
  }
}

// ---------------------------------------------------------------------------
extern "C" void kernel_launch(void* const* d_in, const int* in_sizes, int n_in,
                              void* d_out, int out_size, void* d_ws,
                              size_t ws_size, hipStream_t stream) {
  const float* x  = (const float*)d_in[0];
  const float* Wq = (const float*)d_in[1];
  const float* bq = (const float*)d_in[2];
  const float* Wk = (const float*)d_in[3];
  const float* bk = (const float*)d_in[4];
  const float* Wv = (const float*)d_in[5];
  const float* bv = (const float*)d_in[6];
  const float* Wo = (const float*)d_in[7];
  const float* bo = (const float*)d_in[8];
  const unsigned char* pad = (const unsigned char*)d_in[9];
  const int* causal = (const int*)d_in[10];

  // Workspace: Q, K^T, V (bf16) + ctx (B,T,D) bf16 = 32 MB
  const size_t NE = (size_t)B_ * H_ * T_ * DH_;  // 4,194,304
  bf16* Q   = (bf16*)d_ws;
  bf16* KT  = Q + NE;
  bf16* V   = KT + NE;
  bf16* CTX = V + NE;

  qkv_gemm<<<dim3((B_ * T_) / 128, D_ / 128, 3), 256, 0, stream>>>(
      x, Wq, bq, Wk, bk, Wv, bv, Q, KT, V);

  const size_t attn_lds = (size_t)16 * 2048 * 4 + 4 * 8 * 32 * 4;
  attn_topk<<<dim3(T_ / 16, H_, B_), 256, attn_lds, stream>>>(
      Q, KT, V, pad, causal, CTX);

  out_gemm<<<dim3((B_ * T_) / 128, D_ / 128, 1), 256, 0, stream>>>(
      CTX, Wo, bo, (float*)d_out);
}
